// SimpleCrossAttention_12730283065981
// MI455X (gfx1250) — matrix operand order repaired and checked by reading, not compiled, not run on hardware
//
#include <hip/hip_runtime.h>

typedef __attribute__((ext_vector_type(16))) __bf16 v16bf;
typedef __attribute__((ext_vector_type(8)))  float  v8f;

#define N_ROWS 50000
#define DIM    512
#define HDIM   128
#define TILES  (N_ROWS / 16)       // 3125 exactly
#define WPB    4                   // waves per block in rows_kernel

union Frag16 { uint4 q[2]; v16bf v; };

static __device__ inline unsigned pack_bf16x2(float a, float b) {
    union { __bf16 h[2]; unsigned u; } t;
    t.h[0] = (__bf16)a; t.h[1] = (__bf16)b;
    return t.u;
}

// ---------------- kernel 1: W (f32, [H,D]) -> bf16 ----------------
__global__ void convert_w_kernel(const float* __restrict__ w1,
                                 const float* __restrict__ w2,
                                 __bf16* __restrict__ wh) {
    int i = blockIdx.x * blockDim.x + threadIdx.x;
    const int n = HDIM * DIM;
    if (i < n)            wh[i] = (__bf16)w1[i];
    else if (i < 2 * n)   wh[i] = (__bf16)w2[i - n];
}

// -------- per-wave: layernorm 16 rows, stage bf16 into LDS --------
static __device__ inline void norm_stage(const float* __restrict__ x, int r0, int lane,
                                         const float* __restrict__ nw,
                                         const float* __restrict__ nb,
                                         unsigned* __restrict__ stage_u) {
    for (int m = 0; m < 16; ++m) {
        const float4* row = (const float4*)(x + (size_t)(r0 + m) * DIM);
        float4 v[4];
        float s = 0.f, ss = 0.f;
#pragma unroll
        for (int i = 0; i < 4; ++i) {
            v[i] = row[i * 32 + lane];
            s  += v[i].x + v[i].y + v[i].z + v[i].w;
            ss += v[i].x * v[i].x + v[i].y * v[i].y + v[i].z * v[i].z + v[i].w * v[i].w;
        }
#pragma unroll
        for (int off = 16; off >= 1; off >>= 1) {
            s  += __shfl_xor(s,  off, 32);
            ss += __shfl_xor(ss, off, 32);
        }
        const float inv = 1.0f / 512.0f;
        float mu  = s * inv;
        float var = ss * inv - mu * mu;
        float rs  = rsqrtf(var + 1e-5f);
#pragma unroll
        for (int i = 0; i < 4; ++i) {
            int d0 = i * 128 + lane * 4;
            float y0 = (v[i].x - mu) * rs * nw[d0 + 0] + nb[d0 + 0];
            float y1 = (v[i].y - mu) * rs * nw[d0 + 1] + nb[d0 + 1];
            float y2 = (v[i].z - mu) * rs * nw[d0 + 2] + nb[d0 + 2];
            float y3 = (v[i].w - mu) * rs * nw[d0 + 3] + nb[d0 + 3];
            unsigned idx = (unsigned)(m * DIM + d0) >> 1;   // dword index
            stage_u[idx + 0] = pack_bf16x2(y0, y1);
            stage_u[idx + 1] = pack_bf16x2(y2, y3);
        }
    }
    __threadfence_block();   // order LDS stores before the DS fragment loads
}

// -------- per-wave GEMM: [16x512](LDS bf16) x W^T -> h[16x128] f32 --------
static __device__ inline void gemm_tile(const __bf16* __restrict__ stage,
                                        const __bf16* __restrict__ wh,
                                        const float* __restrict__ bias_lds,
                                        int lane, v8f acc[8]) {
    const v8f z = {0.f, 0.f, 0.f, 0.f, 0.f, 0.f, 0.f, 0.f};
#pragma unroll
    for (int ct = 0; ct < 8; ++ct) acc[ct] = z;
    const int mrow = lane & 15;
    const int half = lane >> 4;
    for (int kb = 0; kb < 16; ++kb) {
        // A fragment (16-bit A 16x32): lanes 0-15: K{0..7,16..23}; 16-31: K{8..15,24..31}
        Frag16 A;
        const uint4* ap = (const uint4*)(stage + mrow * DIM + kb * 32 + half * 8);
        A.q[0] = ap[0];
        A.q[1] = ap[2];          // +16 halfs = +32 bytes
#pragma unroll
        for (int ct = 0; ct < 8; ++ct) {
            // B fragment (16-bit B 32x16): col = lane%16; lanes 0-15: K{0..7,16..23}; 16-31: K{8..15,24..31}
            Frag16 B;
            const uint4* bp = (const uint4*)(wh + (size_t)(ct * 16 + mrow) * DIM
                                                + kb * 32 + half * 8);
            B.q[0] = bp[0];
            B.q[1] = bp[2];
            acc[ct] = __builtin_amdgcn_wmma_f32_16x16x32_bf16(
                false, A.v, false, B.v, (short)0, acc[ct], false, false);
        }
    }
#pragma unroll
    for (int ct = 0; ct < 8; ++ct) {
        float b = bias_lds[ct * 16 + mrow];   // C layout: column N = lane%16
#pragma unroll
        for (int j = 0; j < 8; ++j) acc[ct][j] += b;
    }
}

// ---------------- kernel 2: LN + 2 GEMMs + cosine per row ----------------
__global__ __launch_bounds__(128)
void rows_kernel(const float* __restrict__ x1, const float* __restrict__ x2,
                 const float* __restrict__ nw_g, const float* __restrict__ nb_g,
                 const __bf16* __restrict__ w1h, const __bf16* __restrict__ w2h,
                 const float* __restrict__ b1_g, const float* __restrict__ b2_g,
                 float* __restrict__ cosv) {
    extern __shared__ unsigned char smem_raw[];
    float*  nw     = (float*)smem_raw;       // 512
    float*  nb     = nw + DIM;               // 512
    float*  bb1    = nb + DIM;               // 128
    float*  bb2    = bb1 + HDIM;             // 128
    __bf16* stage0 = (__bf16*)(bb2 + HDIM);  // WPB * 16*512 bf16

    const int tid = threadIdx.x;
    for (int i = tid; i < DIM; i += 128) { nw[i] = nw_g[i]; nb[i] = nb_g[i]; }
    if (tid < HDIM) { bb1[tid] = b1_g[tid]; bb2[tid] = b2_g[tid]; }
    __syncthreads();

    const int wave = tid >> 5;
    const int lane = tid & 31;
    const int tile = blockIdx.x * WPB + wave;
    if (tile >= TILES) return;               // whole wave exits -> EXEC all-1s below
    const int r0 = tile * 16;
    __bf16*   stage   = stage0 + wave * (16 * DIM);
    unsigned* stage_u = (unsigned*)stage;

    v8f h1[8], h2[8];
    norm_stage(x1, r0, lane, nw, nb, stage_u);
    gemm_tile(stage, w1h, bb1, lane, h1);
    norm_stage(x2, r0, lane, nw, nb, stage_u);
    gemm_tile(stage, w2h, bb2, lane, h2);

    // per-row dot / norms: row m = j + 8*(lane/16); columns striped over lane%16 and ct
    float pd[8], p1[8], p2[8];
#pragma unroll
    for (int j = 0; j < 8; ++j) { pd[j] = 0.f; p1[j] = 0.f; p2[j] = 0.f; }
#pragma unroll
    for (int ct = 0; ct < 8; ++ct)
#pragma unroll
        for (int j = 0; j < 8; ++j) {
            float a = h1[ct][j], b = h2[ct][j];
            pd[j] += a * b; p1[j] += a * a; p2[j] += b * b;
        }
#pragma unroll
    for (int off = 8; off >= 1; off >>= 1)
#pragma unroll
        for (int j = 0; j < 8; ++j) {
            pd[j] += __shfl_xor(pd[j], off, 16);
            p1[j] += __shfl_xor(p1[j], off, 16);
            p2[j] += __shfl_xor(p2[j], off, 16);
        }
    if ((lane & 15) == 0) {
        int mb = (lane >> 4) * 8;
#pragma unroll
        for (int j = 0; j < 8; ++j) {
            float n1 = fmaxf(sqrtf(p1[j]), 1e-6f);
            float n2 = fmaxf(sqrtf(p2[j]), 1e-6f);
            cosv[r0 + mb + j] = pd[j] / (n1 * n2);
        }
    }
}

// ---------------- kernel 3: softmax max & sum; zero s-vectors ----------------
__global__ __launch_bounds__(1024)
void softmax_stats_kernel(const float* __restrict__ cosv,
                          float* __restrict__ stats, float* __restrict__ svec) {
    __shared__ float red[1024];
    int t = threadIdx.x;
    float m = -3.402823466e38f;
    for (int i = t; i < N_ROWS; i += 1024) m = fmaxf(m, cosv[i]);
    red[t] = m; __syncthreads();
    for (int s = 512; s > 0; s >>= 1) {
        if (t < s) red[t] = fmaxf(red[t], red[t + s]);
        __syncthreads();
    }
    float mx = red[0];
    __syncthreads();
    float sum = 0.f;
    for (int i = t; i < N_ROWS; i += 1024) sum += expf(cosv[i] - mx);
    red[t] = sum; __syncthreads();
    for (int s = 512; s > 0; s >>= 1) {
        if (t < s) red[t] += red[t + s];
        __syncthreads();
    }
    if (t == 0) { stats[0] = mx; stats[1] = red[0]; }
    svec[t] = 0.f;   // zero s1(512) || s2(512) every call (graph-replay safe)
}

// ---------------- kernel 4: sim = softmax(cos) ----------------
__global__ void sim_kernel(const float* __restrict__ cosv,
                           const float* __restrict__ stats, float* __restrict__ sim) {
    int i = blockIdx.x * blockDim.x + threadIdx.x;
    if (i < N_ROWS) sim[i] = expf(cosv[i] - stats[0]) / stats[1];
}

// ---------------- kernel 5: s1 = sim@x1, s2 = sim@x2 ----------------
#define WSUM_BLOCKS 200
#define ROWS_PER_BLK 250
__global__ __launch_bounds__(256)
void wsum_kernel(const float* __restrict__ x1, const float* __restrict__ x2,
                 const float* __restrict__ sim, float* __restrict__ svec) {
    int t = threadIdx.x;
    int rbeg = blockIdx.x * ROWS_PER_BLK;
    int rend = rbeg + ROWS_PER_BLK; if (rend > N_ROWS) rend = N_ROWS;
    float a0 = 0.f, a1 = 0.f, b0 = 0.f, b1 = 0.f;
    for (int i = rbeg; i < rend; ++i) {
        float w = sim[i];
        size_t base = (size_t)i * DIM;
        a0 += w * x1[base + t];
        a1 += w * x1[base + t + 256];
        b0 += w * x2[base + t];
        b1 += w * x2[base + t + 256];
    }
    unsafeAtomicAdd(&svec[t],             a0);
    unsafeAtomicAdd(&svec[t + 256],       a1);
    unsafeAtomicAdd(&svec[512 + t],       b0);
    unsafeAtomicAdd(&svec[512 + t + 256], b1);
}

// ---------------- kernel 6: out = x + s (broadcast), float4 ----------------
__global__ __launch_bounds__(256)
void add_kernel(const float* __restrict__ x1, const float* __restrict__ x2,
                const float* __restrict__ svec, float* __restrict__ out) {
    size_t idx = (size_t)blockIdx.x * blockDim.x + threadIdx.x;  // float4 index
    const size_t per = (size_t)N_ROWS * (DIM / 4);
    if (idx >= 2 * per) return;
    int half = idx >= per;
    size_t j = idx - (half ? per : 0);
    int c4 = (int)(j & (DIM / 4 - 1));
    const float4* xs = (const float4*)(half ? x2 : x1);
    const float4* sv = (const float4*)svec + (half ? (DIM / 4) : 0);
    float4 v = xs[j];
    float4 s = sv[c4];
    float4 o = { v.x + s.x, v.y + s.y, v.z + s.z, v.w + s.w };
    ((float4*)out)[idx] = o;
}

extern "C" void kernel_launch(void* const* d_in, const int* in_sizes, int n_in,
                              void* d_out, int out_size, void* d_ws, size_t ws_size,
                              hipStream_t stream) {
    const float* x1 = (const float*)d_in[0];
    const float* x2 = (const float*)d_in[1];
    const float* nw = (const float*)d_in[2];
    const float* nb = (const float*)d_in[3];
    const float* w1 = (const float*)d_in[4];
    const float* b1 = (const float*)d_in[5];
    const float* w2 = (const float*)d_in[6];
    const float* b2 = (const float*)d_in[7];
    float* out = (float*)d_out;

    unsigned char* ws = (unsigned char*)d_ws;
    __bf16* wh    = (__bf16*)ws;                 // 2*H*D bf16 = 262144 B
    float*  cosv  = (float*)(ws + 262144);       // 200000 B
    float*  sim   = (float*)(ws + 462848);       // 200000 B
    float*  stats = (float*)(ws + 663552);       // 8 B
    float*  svec  = (float*)(ws + 663680);       // 4096 B (s1 || s2)

    convert_w_kernel<<<(2 * HDIM * DIM + 255) / 256, 256, 0, stream>>>(w1, w2, wh);

    size_t lds = 2 * DIM * sizeof(float) + 2 * HDIM * sizeof(float)
               + (size_t)WPB * 16 * DIM * sizeof(__bf16);   // 70656 B
    int blocks = (TILES + WPB - 1) / WPB;                   // 782
    rows_kernel<<<blocks, 128, lds, stream>>>(x1, x2, nw, nb,
                                              wh, wh + HDIM * DIM, b1, b2, cosv);

    softmax_stats_kernel<<<1, 1024, 0, stream>>>(cosv, stats, svec);
    sim_kernel<<<(N_ROWS + 255) / 256, 256, 0, stream>>>(cosv, stats, sim);
    wsum_kernel<<<WSUM_BLOCKS, 256, 0, stream>>>(x1, x2, sim, svec);
    add_kernel<<<(int)((2 * (size_t)N_ROWS * (DIM / 4) + 255) / 256), 256, 0, stream>>>(
        x1, x2, svec, out);
}
